// CompletionNetSmaller_58677843198409
// MI455X (gfx1250) — matrix operand order repaired and checked by reading, not compiled, not run on hardware
//
#include <hip/hip_runtime.h>
#include <cmath>

typedef _Float16 v16h __attribute__((ext_vector_type(16)));
typedef float    v8f  __attribute__((ext_vector_type(8)));
typedef float    f4   __attribute__((ext_vector_type(4)));
typedef unsigned int u32x4 __attribute__((ext_vector_type(4)));
typedef int          i32x8 __attribute__((ext_vector_type(8)));
typedef int          i32x4 __attribute__((ext_vector_type(4)));

#if __has_builtin(__builtin_amdgcn_tensor_load_to_lds) && __has_builtin(__builtin_amdgcn_s_wait_tensorcnt)
#define HAVE_TDM 1
#else
#define HAVE_TDM 0
#endif

// Weight tile staged per GEMM step: 32 rows (cin) x 128 cols (cout) fp32,
// LDS row stride 129 floats (TDM pad: 1 DWORD after every 128 DWORDs).
#define TILE_B     (32 * 129)                              // floats per buffer
#define SMEM_SCONV (2 * TILE_B * 4 + 16 * 34 * 2 + 64)     // 34176 bytes
#define SMEM_DEEP  (2 * TILE_B * 4 + 128 * 34 * 2 + 512)   // 42240 bytes

#if HAVE_TDM
// Issue a TDM 2D tile load: gsrc points at tile start, row stride = Cout
// elements, 128 cols x 32 rows; rows beyond rowsValid are hardware zero-fill.
__device__ __forceinline__ void tdm_load_tile(const float* gsrc, unsigned ldsOff,
                                              int rowsValid, int Cout)
{
    unsigned long long ga = (unsigned long long)(uintptr_t)gsrc;
    u32x4 g0;
    g0.x = 1u;                                            // count=1 (user D#)
    g0.y = ldsOff;                                        // lds_addr (bytes)
    g0.z = (unsigned)ga;                                  // global_addr[31:0]
    g0.w = (unsigned)((ga >> 32) & 0x01ffffffu) | (2u << 30);  // addr[56:32] | type=2
    if (rowsValid < 0) rowsValid = 0;
    i32x8 g1;
    g1[0] = (2 << 16) | (1 << 20) | (6 << 22);            // data_size=4B, pad_en, 1DW per 128DW
    g1[1] = (int)(128u << 16);                            // tensor_dim0 = 128 (lo16 in [31:16])
    g1[2] = (int)(((unsigned)rowsValid & 0xffffu) << 16); // tensor_dim0 hi=0 | tensor_dim1 lo
    g1[3] = (int)(128u << 16);                            // tensor_dim1 hi=0 | tile_dim0 = 128
    g1[4] = 32;                                           // tile_dim1 = 32, tile_dim2 = 0
    g1[5] = Cout;                                         // tensor_dim0_stride[31:0] (elements)
    g1[6] = 0;                                            // stride hi | dim1_stride lo (unused, 2D)
    g1[7] = 0;
    i32x4 gz4 = {0, 0, 0, 0};                             // groups 2/3 unused (<=2D)
    i32x8 gz8 = {0, 0, 0, 0, 0, 0, 0, 0};
    __builtin_amdgcn_tensor_load_to_lds(g0, g1, gz4, gz4, gz8, 0);
}
#endif

// ---------------------------------------------------------------------------
// Fused gather + GEMM sparse conv:  Y[Nout,Cout] = sum_k gather(X, nbr[k]) @ W[k]
// 256 threads = 8 waves; tile = 16 points x 128 cout (one 16x16 WMMA per wave).
// ---------------------------------------------------------------------------
__global__ void sconv_wmma_f16(const float* __restrict__ X,
                               const float* __restrict__ W,
                               const int*   __restrict__ nbr,
                               float* __restrict__ Y,
                               int Nout, int Cin, int Cout, int K)
{
    extern __shared__ char smem[];
    float*    sB   = (float*)smem;                          // [2][TILE_B]
    _Float16* sA   = (_Float16*)(smem + 2 * TILE_B * 4);    // [16][34]
    int*      sIdx = (int*)(smem + 2 * TILE_B * 4 + 16 * 34 * 2);

    const int tid  = threadIdx.x;
    const int lane = tid & 31;
    const int wid  = tid >> 5;
    const int half = lane >> 4;
    const int l16  = lane & 15;

    const int i0  = blockIdx.x * 16;
    const int j0  = blockIdx.y * 128;
    const int col = j0 + wid * 16 + l16;
    const int nc  = (Cin + 31) >> 5;
    const int nq  = K * nc;
    const int vecA = ((Cin & 31) == 0);

    v8f acc = {};

#if HAVE_TDM
    if (wid == 0) {   // chunk 0: k=0, c0=0
        int rows = Cin < 32 ? Cin : 32;
        tdm_load_tile(W + j0, (unsigned)(uintptr_t)sB, rows, Cout);
    }
#endif

    for (int q = 0; q < nq; ++q) {
        const int k   = q / nc;
        const int c0  = (q - k * nc) << 5;
        const int buf = q & 1;
        float* sBc = sB + buf * TILE_B;

        __syncthreads();   // prior reads of sA and sB[1-buf] complete
        if (tid < 16) {
            int row = i0 + tid;
            sIdx[tid] = (row < Nout) ? nbr[(size_t)k * Nout + row] : -1;
        }
#if HAVE_TDM
        if (wid == 0 && q + 1 < nq) {
            int k2 = (q + 1) / nc;
            int c2 = ((q + 1) - k2 * nc) << 5;
            int rows = Cin - c2; if (rows > 32) rows = 32;
            tdm_load_tile(W + (size_t)k2 * Cin * Cout + (size_t)c2 * Cout + j0,
                          (unsigned)(uintptr_t)(sB + (1 - buf) * TILE_B), rows, Cout);
        }
#endif
        __syncthreads();   // sIdx visible

        if (vecA) {
            for (int e = tid; e < 128; e += 256) {          // 16 rows x 8 xfloat4
                int r = e >> 3, cc = (e & 7) << 2;
                int id = sIdx[r];
                f4 v = {0.f, 0.f, 0.f, 0.f};
                if (id >= 0) v = *(const f4*)(X + (size_t)id * Cin + c0 + cc);
                sA[r * 34 + cc]     = (_Float16)v.x;
                sA[r * 34 + cc + 1] = (_Float16)v.y;
                sA[r * 34 + cc + 2] = (_Float16)v.z;
                sA[r * 34 + cc + 3] = (_Float16)v.w;
            }
        } else {
            for (int e = tid; e < 512; e += 256) {
                int r = e >> 5, cc = e & 31;
                int id = sIdx[r];
                int ci = c0 + cc;
                float v = (id >= 0 && ci < Cin) ? X[(size_t)id * Cin + ci] : 0.f;
                sA[r * 34 + cc] = (_Float16)v;
            }
        }

#if HAVE_TDM
        if (wid == 0) {    // chunk q landed (TDM in-order per wave)
            if (q + 1 < nq) __builtin_amdgcn_s_wait_tensorcnt(1);
            else            __builtin_amdgcn_s_wait_tensorcnt(0);
        }
#else
        for (int e = tid; e < 32 * 128; e += 256) {          // sync fallback stage of B
            int r = e >> 7, cc = e & 127;
            int ci = c0 + r;
            sBc[r * 129 + cc] =
                (ci < Cin) ? W[(size_t)k * Cin * Cout + (size_t)ci * Cout + j0 + cc] : 0.f;
        }
#endif
        __syncthreads();   // sA staged + sB[buf] ready for all 8 waves

        v16h a, b;
#pragma unroll
        for (int j = 0; j < 8; ++j) {
            int kb = half * 16 + 2 * j;
            a[2 * j]     = sA[l16 * 34 + kb];
            a[2 * j + 1] = sA[l16 * 34 + kb + 1];
        }
        const int kbase = half * 16;
#pragma unroll
        for (int j = 0; j < 8; ++j) {
            int kk = kbase + 2 * j;
            int ci = c0 + kk;
            float w0 = sBc[kk * 129 + wid * 16 + l16];
            float w1 = sBc[(kk + 1) * 129 + wid * 16 + l16];
            b[2 * j]     = (ci     < Cin) ? (_Float16)w0 : (_Float16)0.f;
            b[2 * j + 1] = (ci + 1 < Cin) ? (_Float16)w1 : (_Float16)0.f;
        }
        acc = __builtin_amdgcn_wmma_f32_16x16x32_f16(false, a, false, b,
                                                     (short)0, acc, false, false);
    }

#pragma unroll
    for (int r = 0; r < 8; ++r) {
        int row = i0 + r + half * 8;
        if (row < Nout) Y[(size_t)row * Cout + col] = acc[r];
    }
}

// ---------------------------------------------------------------------------
// Deep-row split-K variant for 16 < Nout <= 128: grid = (Cout/128, K).
// All 8 row tiles accumulate in registers; each staged weight chunk is read
// from HBM exactly once and reused by 8 WMMAs sharing one B fragment.
// Partial sums over k land via global_atomic_add_f32 into pre-zeroed Y.
// ---------------------------------------------------------------------------
__global__ void sconv_deep_wmma(const float* __restrict__ X,
                                const float* __restrict__ W,
                                const int*   __restrict__ nbr,
                                float* __restrict__ Y,
                                int Nout, int Cin, int Cout)
{
    extern __shared__ char smem[];
    float*    sB   = (float*)smem;                          // [2][TILE_B]
    _Float16* sA   = (_Float16*)(smem + 2 * TILE_B * 4);    // [128][34]
    int*      sIdx = (int*)(smem + 2 * TILE_B * 4 + 128 * 34 * 2);

    const int tid  = threadIdx.x;
    const int lane = tid & 31;
    const int wid  = tid >> 5;
    const int half = lane >> 4;
    const int l16  = lane & 15;

    const int j0  = blockIdx.x * 128;
    const int k   = blockIdx.y;
    const int col = j0 + wid * 16 + l16;
    const int nc  = (Cin + 31) >> 5;
    const int vecA = ((Cin & 31) == 0);
    const float* Wk = W + (size_t)k * Cin * Cout + j0;

    if (tid < 128) sIdx[tid] = (tid < Nout) ? nbr[(size_t)k * Nout + tid] : -1;

    v8f acc[8] = {};

#if HAVE_TDM
    if (wid == 0) {
        int rows = Cin < 32 ? Cin : 32;
        tdm_load_tile(Wk, (unsigned)(uintptr_t)sB, rows, Cout);
    }
#endif

    for (int q = 0; q < nc; ++q) {
        const int c0  = q << 5;
        const int buf = q & 1;
        float* sBc = sB + buf * TILE_B;

        __syncthreads();   // sIdx visible (q=0); prior reads of sA/sB[1-buf] done
#if HAVE_TDM
        if (wid == 0 && q + 1 < nc) {
            int c2 = (q + 1) << 5;
            int rows = Cin - c2; if (rows > 32) rows = 32;
            tdm_load_tile(Wk + (size_t)c2 * Cout,
                          (unsigned)(uintptr_t)(sB + (1 - buf) * TILE_B), rows, Cout);
        }
#endif
        // stage A for all 128 rows (8 row tiles x 16) as f16
        if (vecA) {
            for (int e = tid; e < 1024; e += 256) {         // 128 rows x 8 xfloat4
                int r = e >> 3, cc = (e & 7) << 2;
                int id = sIdx[r];
                f4 v = {0.f, 0.f, 0.f, 0.f};
                if (id >= 0) v = *(const f4*)(X + (size_t)id * Cin + c0 + cc);
                sA[r * 34 + cc]     = (_Float16)v.x;
                sA[r * 34 + cc + 1] = (_Float16)v.y;
                sA[r * 34 + cc + 2] = (_Float16)v.z;
                sA[r * 34 + cc + 3] = (_Float16)v.w;
            }
        } else {
            for (int e = tid; e < 4096; e += 256) {
                int r = e >> 5, cc = e & 31;
                int id = sIdx[r];
                int ci = c0 + cc;
                float v = (id >= 0 && ci < Cin) ? X[(size_t)id * Cin + ci] : 0.f;
                sA[r * 34 + cc] = (_Float16)v;
            }
        }

#if HAVE_TDM
        if (wid == 0) {
            if (q + 1 < nc) __builtin_amdgcn_s_wait_tensorcnt(1);
            else            __builtin_amdgcn_s_wait_tensorcnt(0);
        }
#else
        for (int e = tid; e < 32 * 128; e += 256) {
            int r = e >> 7, cc = e & 127;
            int ci = c0 + r;
            sBc[r * 129 + cc] = (ci < Cin) ? Wk[(size_t)ci * Cout + cc] : 0.f;
        }
#endif
        __syncthreads();

        v16h b;
        const int kbase = half * 16;
#pragma unroll
        for (int j = 0; j < 8; ++j) {
            int kk = kbase + 2 * j;
            int ci = c0 + kk;
            float w0 = sBc[kk * 129 + wid * 16 + l16];
            float w1 = sBc[(kk + 1) * 129 + wid * 16 + l16];
            b[2 * j]     = (ci     < Cin) ? (_Float16)w0 : (_Float16)0.f;
            b[2 * j + 1] = (ci + 1 < Cin) ? (_Float16)w1 : (_Float16)0.f;
        }
#pragma unroll
        for (int rt = 0; rt < 8; ++rt) {
            v16h a;
            const int rb = (rt * 16 + l16) * 34;
#pragma unroll
            for (int j = 0; j < 8; ++j) {
                int kb = half * 16 + 2 * j;
                a[2 * j]     = sA[rb + kb];
                a[2 * j + 1] = sA[rb + kb + 1];
            }
            acc[rt] = __builtin_amdgcn_wmma_f32_16x16x32_f16(false, a, false, b,
                                                             (short)0, acc[rt], false, false);
        }
    }

#pragma unroll
    for (int rt = 0; rt < 8; ++rt) {
#pragma unroll
        for (int r = 0; r < 8; ++r) {
            int row = rt * 16 + r + half * 8;
            if (row < Nout) atomicAdd(&Y[(size_t)row * Cout + col], acc[rt][r]);
        }
    }
}

// ---------------------------------------------------------------------------
// Generative transpose conv k=2 s=2: Y[k*Np + p, :] = X[p,:] @ W[k]
// grid.z = child offset k. Same TDM-staged weight stream, no gather map.
// ---------------------------------------------------------------------------
__global__ void gtconv_wmma_f16(const float* __restrict__ X,
                                const float* __restrict__ W,
                                float* __restrict__ Y,
                                int Np, int Cin, int Cout)
{
    extern __shared__ char smem[];
    float*    sB = (float*)smem;
    _Float16* sA = (_Float16*)(smem + 2 * TILE_B * 4);

    const int tid  = threadIdx.x;
    const int lane = tid & 31;
    const int wid  = tid >> 5;
    const int half = lane >> 4;
    const int l16  = lane & 15;

    const int k   = blockIdx.z;
    const int p0  = blockIdx.x * 16;
    const int j0  = blockIdx.y * 128;
    const int col = j0 + wid * 16 + l16;
    const int nc  = (Cin + 31) >> 5;
    const float* Wk = W + (size_t)k * Cin * Cout + j0;

    v8f acc = {};

#if HAVE_TDM
    if (wid == 0) {
        int rows = Cin < 32 ? Cin : 32;
        tdm_load_tile(Wk, (unsigned)(uintptr_t)sB, rows, Cout);
    }
#endif

    for (int q = 0; q < nc; ++q) {
        const int c0  = q << 5;
        const int buf = q & 1;
        float* sBc = sB + buf * TILE_B;

        __syncthreads();
#if HAVE_TDM
        if (wid == 0 && q + 1 < nc) {
            int c2 = (q + 1) << 5;
            int rows = Cin - c2; if (rows > 32) rows = 32;
            tdm_load_tile(Wk + (size_t)c2 * Cout,
                          (unsigned)(uintptr_t)(sB + (1 - buf) * TILE_B), rows, Cout);
        }
#endif
        for (int e = tid; e < 128; e += 256) {               // 16 rows x 8 xfloat4
            int r = e >> 3, cc = (e & 7) << 2;
            int p = p0 + r;
            f4 v = {0.f, 0.f, 0.f, 0.f};
            if (p < Np && (Cin & 3) == 0)
                v = *(const f4*)(X + (size_t)p * Cin + c0 + cc);
            sA[r * 34 + cc]     = (_Float16)v.x;
            sA[r * 34 + cc + 1] = (_Float16)v.y;
            sA[r * 34 + cc + 2] = (_Float16)v.z;
            sA[r * 34 + cc + 3] = (_Float16)v.w;
        }
#if HAVE_TDM
        if (wid == 0) {
            if (q + 1 < nc) __builtin_amdgcn_s_wait_tensorcnt(1);
            else            __builtin_amdgcn_s_wait_tensorcnt(0);
        }
#else
        for (int e = tid; e < 32 * 128; e += 256) {
            int r = e >> 7, cc = e & 127;
            int ci = c0 + r;
            sBc[r * 129 + cc] = (ci < Cin) ? Wk[(size_t)ci * Cout + cc] : 0.f;
        }
#endif
        __syncthreads();

        v16h a, b;
#pragma unroll
        for (int j = 0; j < 8; ++j) {
            int kb = half * 16 + 2 * j;
            a[2 * j]     = sA[l16 * 34 + kb];
            a[2 * j + 1] = sA[l16 * 34 + kb + 1];
        }
        const int kbase = half * 16;
#pragma unroll
        for (int j = 0; j < 8; ++j) {
            int kk = kbase + 2 * j;
            int ci = c0 + kk;
            float w0 = sBc[kk * 129 + wid * 16 + l16];
            float w1 = sBc[(kk + 1) * 129 + wid * 16 + l16];
            b[2 * j]     = (ci     < Cin) ? (_Float16)w0 : (_Float16)0.f;
            b[2 * j + 1] = (ci + 1 < Cin) ? (_Float16)w1 : (_Float16)0.f;
        }
        acc = __builtin_amdgcn_wmma_f32_16x16x32_f16(false, a, false, b,
                                                     (short)0, acc, false, false);
    }

#pragma unroll
    for (int r = 0; r < 8; ++r) {
        int p = p0 + r + half * 8;
        if (p < Np) Y[((size_t)k * Np + p) * Cout + col] = acc[r];
    }
}

// ---------------------------------------------------------------------------
__global__ void zero_k(float* __restrict__ D, long n)
{
    long i = (long)blockIdx.x * 256 + threadIdx.x;
    if (i < n) D[i] = 0.f;
}

__global__ void bn_stats_k(const float* __restrict__ X, int N, int C,
                           float* __restrict__ mean, float* __restrict__ rstd)
{
    __shared__ float s1[256], s2[256];
    int c = blockIdx.x;
    float a = 0.f, b = 0.f;
    for (int r = threadIdx.x; r < N; r += 256) {
        float v = X[(size_t)r * C + c];
        a += v; b += v * v;
    }
    s1[threadIdx.x] = a; s2[threadIdx.x] = b;
    __syncthreads();
    for (int s = 128; s > 0; s >>= 1) {
        if (threadIdx.x < s) {
            s1[threadIdx.x] += s1[threadIdx.x + s];
            s2[threadIdx.x] += s2[threadIdx.x + s];
        }
        __syncthreads();
    }
    if (threadIdx.x == 0) {
        float m = s1[0] / (float)N;
        float v = s2[0] / (float)N - m * m;
        mean[c] = m;
        rstd[c] = rsqrtf(v + 1e-5f);
    }
}

__global__ void bn_elu_k(const float* __restrict__ X, const float* __restrict__ gb,
                         const float* __restrict__ mean, const float* __restrict__ rstd,
                         float* __restrict__ Y, int N, int C)
{
    size_t i = (size_t)blockIdx.x * 256 + threadIdx.x;
    if (i >= (size_t)N * C) return;
    int c = (int)(i % C);
    float z = gb[c] * (X[i] - mean[c]) * rstd[c] + gb[C + c];
    Y[i] = z > 0.f ? z : (expf(z) - 1.f);
}

__global__ void skip_add_k(float* __restrict__ D, const float* __restrict__ E,
                           const int* __restrict__ map, int Ne, int C)
{
    size_t i = (size_t)blockIdx.x * 256 + threadIdx.x;
    if (i >= (size_t)Ne * C) return;
    int r = (int)(i / C), c = (int)(i % C);
    int d = map[r];
    if (d >= 0) D[(size_t)d * C + c] += E[i];
}

__global__ void cls_head_k(const float* __restrict__ X, const float* __restrict__ Wc,
                           const float* __restrict__ Bc, float* __restrict__ Y,
                           int N, int C, int O)
{
    int i = blockIdx.x * 256 + threadIdx.x;
    if (i >= N * O) return;
    int n = i / O, o = i % O;
    const float* x = X + (size_t)n * C;
    float s = Bc[o];
    for (int c = 0; c < C; ++c) s += x[c] * Wc[(size_t)c * O + o];
    Y[i] = s;
}

__global__ void target_mask_k(const int* __restrict__ coords, const int* __restrict__ tgt,
                              int N, int Ntgt, int stride, float* __restrict__ mask)
{
    int n = blockIdx.x * 256 + threadIdx.x;
    if (n >= N) return;
    const int* c = coords + (size_t)n * 4;
    int key = ((c[0] * 64 + c[1]) * 64 + c[2]) * 64 + c[3];
    float hit = 0.f;
    for (int t = 0; t < Ntgt; ++t) {
        const int* q = tgt + (size_t)t * 4;
        int tk = ((q[0] * 64 + q[1] / stride) * 64 + q[2] / stride) * 64 + q[3] / stride;
        if (tk == key) hit = 1.f;
    }
    mask[n] = hit;
}

__global__ void prune_k(const float* __restrict__ S, float* __restrict__ D,
                        const float* __restrict__ cls, const float* __restrict__ tgt,
                        int N, int C, int O, int useTgt)
{
    size_t i = (size_t)blockIdx.x * 256 + threadIdx.x;
    if (i >= (size_t)N * C) return;
    int n = (int)(i / C);
    bool keep = cls[(size_t)n * O] > 0.f;
    if (useTgt) keep = keep || (tgt[n] > 0.5f);
    D[i] = keep ? S[i] : 0.f;
}

// ---------------------------------------------------------------------------
extern "C" void kernel_launch(void* const* d_in, const int* in_sizes, int n_in,
                              void* d_out, int out_size, void* d_ws, size_t ws_size,
                              hipStream_t stream)
{
    const float* feats   = (const float*)d_in[0];
    const float* w_e1    = (const float*)d_in[1];   const float* bn_e1   = (const float*)d_in[2];
    const float* w_e12a  = (const float*)d_in[3];   const float* bn_e12a = (const float*)d_in[4];
    const float* w_e12b  = (const float*)d_in[5];   const float* bn_e12b = (const float*)d_in[6];
    const float* w_e24a  = (const float*)d_in[7];   const float* bn_e24a = (const float*)d_in[8];
    const float* w_e24b  = (const float*)d_in[9];   const float* bn_e24b = (const float*)d_in[10];
    const float* w_e48a  = (const float*)d_in[11];  const float* bn_e48a = (const float*)d_in[12];
    const float* w_e48b  = (const float*)d_in[13];  const float* bn_e48b = (const float*)d_in[14];
    const float* w_d84a  = (const float*)d_in[15];  const float* bn_d84a = (const float*)d_in[16];
    const float* w_d84b  = (const float*)d_in[17];  const float* bn_d84b = (const float*)d_in[18];
    const float* w_d42a  = (const float*)d_in[19];  const float* bn_d42a = (const float*)d_in[20];
    const float* w_d42b  = (const float*)d_in[21];  const float* bn_d42b = (const float*)d_in[22];
    const float* w_d21a  = (const float*)d_in[23];  const float* bn_d21a = (const float*)d_in[24];
    const float* w_d21b  = (const float*)d_in[25];  const float* bn_d21b = (const float*)d_in[26];
    const float* w_cls4  = (const float*)d_in[27];  const float* b_cls4  = (const float*)d_in[28];
    const float* w_cls2  = (const float*)d_in[29];  const float* b_cls2  = (const float*)d_in[30];
    const float* w_cls1  = (const float*)d_in[31];  const float* b_cls1  = (const float*)d_in[32];
    const int* nbr1   = (const int*)d_in[33];
    const int* nbr2   = (const int*)d_in[34];
    const int* nbr4   = (const int*)d_in[35];
    const int* nbr8   = (const int*)d_in[36];
    const int* nbr_d4 = (const int*)d_in[37];
    const int* nbr_d2 = (const int*)d_in[38];
    const int* nbr_d1 = (const int*)d_in[39];
    const int* dmap12 = (const int*)d_in[40];
    const int* dmap24 = (const int*)d_in[41];
    const int* dmap48 = (const int*)d_in[42];
    const int* e4_in_d4 = (const int*)d_in[43];
    const int* e2_in_d2 = (const int*)d_in[44];
    const int* e1_in_d1 = (const int*)d_in[45];
    const int* coords_d4 = (const int*)d_in[46];
    const int* coords_d2 = (const int*)d_in[47];
    const int* coords_d1 = (const int*)d_in[48];
    const int* tgtc      = (const int*)d_in[49];

    const int N1  = in_sizes[33] / 27;
    const int N2  = in_sizes[34] / 27;
    const int N4  = in_sizes[35] / 27;
    const int N8  = in_sizes[36] / 27;
    const int Nd4 = in_sizes[37] / 27;
    const int Nd2 = in_sizes[38] / 27;
    const int Nd1 = in_sizes[39] / 27;
    const int Ntgt = in_sizes[49] / 4;
    const int F = 15, E0 = 256, E1 = 1024, E2 = 2048, E3 = 2048;

    float* p = (float*)d_ws;
    auto carve = [&](size_t n) { float* r = p; p += ((n + 63) / 64) * 64; return r; };
    float* e1b = carve((size_t)N1 * E0);
    float* e2b = carve((size_t)N2 * E1);
    float* e4b = carve((size_t)N4 * E2);
    float* e8b = carve((size_t)N8 * E3);
    float* d4b = carve((size_t)Nd4 * E2);
    float* d2b = carve((size_t)Nd2 * E1);
    float* d1b = carve((size_t)Nd1 * E0);
    size_t tmax = (size_t)N1 * E0;
    auto mx = [&](size_t v) { if (v > tmax) tmax = v; };
    mx((size_t)N2 * E1); mx((size_t)N4 * E2); mx((size_t)N8 * E3);
    mx((size_t)Nd4 * E2); mx((size_t)Nd2 * E1); mx((size_t)Nd1 * E0);
    float* t     = carve(tmax);
    float* meanb = carve(2048);
    float* rstdb = carve(2048);

    float* out   = (float*)d_out;
    float* ocls4 = out;
    float* ocls2 = ocls4 + Nd4;
    float* ocls1 = ocls2 + Nd2;
    float* otgt4 = ocls1 + (size_t)Nd1 * F;
    float* otgt2 = otgt4 + Nd4;
    float* otgt1 = otgt2 + Nd2;
    float* od1   = otgt1 + Nd1;

    auto cdiv = [](long a, long b) { return (unsigned)((a + b - 1) / b); };

    auto conv = [&](const float* X, const float* W, const int* nb, float* Y,
                    int Nout, int Cin, int Cout, int K) {
        if (Nout > 16 && Nout <= 128) {
            // split-K deep-row path: weights stream from HBM exactly once
            zero_k<<<cdiv((long)Nout * Cout, 256), 256, 0, stream>>>(Y, (long)Nout * Cout);
            dim3 g(Cout / 128, K);
            sconv_deep_wmma<<<g, 256, SMEM_DEEP, stream>>>(X, W, nb, Y, Nout, Cin, Cout);
        } else {
            dim3 g(cdiv(Nout, 16), Cout / 128);
            sconv_wmma_f16<<<g, 256, SMEM_SCONV, stream>>>(X, W, nb, Y, Nout, Cin, Cout, K);
        }
    };
    auto gconv = [&](const float* X, const float* W, float* Y, int Np, int Cin, int Cout) {
        dim3 g(cdiv(Np, 16), Cout / 128, 8);
        gtconv_wmma_f16<<<g, 256, SMEM_SCONV, stream>>>(X, W, Y, Np, Cin, Cout);
    };
    auto bn = [&](const float* Xpre, const float* gb, float* Yact, int N, int C) {
        bn_stats_k<<<C, 256, 0, stream>>>(Xpre, N, C, meanb, rstdb);
        bn_elu_k<<<cdiv((long)N * C, 256), 256, 0, stream>>>(Xpre, gb, meanb, rstdb, Yact, N, C);
    };

    // ---------------- encoder ----------------
    conv(feats, w_e1,   nbr1,   t, N1, F,  E0, 27);  bn(t, bn_e1,   e1b, N1, E0);
    conv(e1b,   w_e12a, dmap12, t, N2, E0, E1, 8);   bn(t, bn_e12a, e2b, N2, E1);
    conv(e2b,   w_e12b, nbr2,   t, N2, E1, E1, 27);  bn(t, bn_e12b, e2b, N2, E1);
    conv(e2b,   w_e24a, dmap24, t, N4, E1, E2, 8);   bn(t, bn_e24a, e4b, N4, E2);
    conv(e4b,   w_e24b, nbr4,   t, N4, E2, E2, 27);  bn(t, bn_e24b, e4b, N4, E2);
    conv(e4b,   w_e48a, dmap48, t, N8, E2, E3, 8);   bn(t, bn_e48a, e8b, N8, E3);
    conv(e8b,   w_e48b, nbr8,   t, N8, E3, E3, 27);  bn(t, bn_e48b, e8b, N8, E3);

    // ---------------- decoder s4 ----------------
    gconv(e8b, w_d84a, t, N8, E3, E2);               bn(t, bn_d84a, d4b, Nd4, E2);
    conv(d4b, w_d84b, nbr_d4, t, Nd4, E2, E2, 27);   bn(t, bn_d84b, d4b, Nd4, E2);
    skip_add_k<<<cdiv((long)N4 * E2, 256), 256, 0, stream>>>(d4b, e4b, e4_in_d4, N4, E2);
    cls_head_k<<<cdiv(Nd4, 256), 256, 0, stream>>>(d4b, w_cls4, b_cls4, ocls4, Nd4, E2, 1);
    target_mask_k<<<cdiv(Nd4, 256), 256, 0, stream>>>(coords_d4, tgtc, Nd4, Ntgt, 4, otgt4);
    prune_k<<<cdiv((long)Nd4 * E2, 256), 256, 0, stream>>>(d4b, d4b, ocls4, otgt4, Nd4, E2, 1, 1);

    // ---------------- decoder s2 ----------------
    gconv(d4b, w_d42a, t, Nd4, E2, E1);              bn(t, bn_d42a, d2b, Nd2, E1);
    conv(d2b, w_d42b, nbr_d2, t, Nd2, E1, E1, 27);   bn(t, bn_d42b, d2b, Nd2, E1);
    skip_add_k<<<cdiv((long)N2 * E1, 256), 256, 0, stream>>>(d2b, e2b, e2_in_d2, N2, E1);
    cls_head_k<<<cdiv(Nd2, 256), 256, 0, stream>>>(d2b, w_cls2, b_cls2, ocls2, Nd2, E1, 1);
    target_mask_k<<<cdiv(Nd2, 256), 256, 0, stream>>>(coords_d2, tgtc, Nd2, Ntgt, 2, otgt2);
    prune_k<<<cdiv((long)Nd2 * E1, 256), 256, 0, stream>>>(d2b, d2b, ocls2, otgt2, Nd2, E1, 1, 1);

    // ---------------- decoder s1 ----------------
    gconv(d2b, w_d21a, t, Nd2, E1, E0);              bn(t, bn_d21a, d1b, Nd1, E0);
    conv(d1b, w_d21b, nbr_d1, t, Nd1, E0, E0, 27);   bn(t, bn_d21b, d1b, Nd1, E0);
    skip_add_k<<<cdiv((long)N1 * E0, 256), 256, 0, stream>>>(d1b, e1b, e1_in_d1, N1, E0);
    cls_head_k<<<cdiv((long)Nd1 * F, 256), 256, 0, stream>>>(d1b, w_cls1, b_cls1, ocls1, Nd1, E0, F);
    target_mask_k<<<cdiv(Nd1, 256), 256, 0, stream>>>(coords_d1, tgtc, Nd1, Ntgt, 1, otgt1);
    prune_k<<<cdiv((long)Nd1 * E0, 256), 256, 0, stream>>>(d1b, od1, ocls1, otgt1, Nd1, E0, F, 0);

    (void)n_in; (void)out_size; (void)ws_size;
}
